// PQLinearBody_47949014892835
// MI455X (gfx1250) — matrix operand order repaired and checked by reading, not compiled, not run on hardware
//
#include <hip/hip_runtime.h>
#include <cstddef>

// ---------------------------------------------------------------------------
// Problem constants (from reference):
//   x:        [4, 2048, 4096] f32  -> flattened [M=8192, K=4096]
//   weight:   [O=2048, I=4096] f32
//   codebook: [256, 8] f32
//   bias:     [2048] f32
//   out:      [4, 2048, 2048] f32  -> flattened [M=8192, N=2048]
// ---------------------------------------------------------------------------
#define O_DIM   2048
#define I_DIM   4096
#define PQ_D    8
#define NCODE   256
#define M_DIM   8192
#define N_DIM   2048
#define K_DIM   4096

#define BM      128
#define BN      128
#define BK      32
#define NKIT    (K_DIM / BK)   // 128 K-steps
#define LDT     40             // padded LDS row stride (ushorts): 80B = 5 banks, 16B-aligned

typedef __attribute__((ext_vector_type(16))) __bf16 v16bf;
typedef __attribute__((ext_vector_type(8)))  float  v8f;

// Exact pointee type the async builtin expects: int __vector_size__(16)
typedef int vint4 __attribute__((vector_size(16)));
typedef void  __attribute__((address_space(1))) as1_void;
typedef void  __attribute__((address_space(3))) as3_void;
typedef vint4 __attribute__((address_space(1))) as1_vint4;
typedef vint4 __attribute__((address_space(3))) as3_vint4;

// CDNA5 async global->LDS path (ASYNCcnt-tracked), with sync fallback.
#if defined(__HIP_DEVICE_COMPILE__)
#if __has_builtin(__builtin_amdgcn_global_load_async_to_lds_b128) && \
    __has_builtin(__builtin_amdgcn_s_wait_asynccnt)
#define USE_ASYNC_LDS 1
#endif
#endif

__device__ __forceinline__ unsigned short f2bf(float f) {
    unsigned int u = __float_as_uint(f);
    u += 0x7FFFu + ((u >> 16) & 1u);   // round-to-nearest-even
    return (unsigned short)(u >> 16);
}

// 16B global -> LDS, per lane.
__device__ __forceinline__ void stage_b16(unsigned short* l, const unsigned short* g) {
#ifdef USE_ASYNC_LDS
    __builtin_amdgcn_global_load_async_to_lds_b128(
        (as1_vint4*)(as1_void*)(void*)(g),
        (as3_vint4*)(as3_void*)(void*)(l), 0, 0);
#else
    *(uint4*)l = *(const uint4*)g;
#endif
}

__device__ __forceinline__ void wait_async_lds() {
#ifdef USE_ASYNC_LDS
    __builtin_amdgcn_s_wait_asynccnt(0);
#endif
}

// ---------------------------------------------------------------------------
// Kernel 1: per-row abs-max scale, clamped at 1e-6   (one block per row)
// ---------------------------------------------------------------------------
__global__ __launch_bounds__(256)
void rowscale_kernel(const float* __restrict__ w, float* __restrict__ scale) {
    __shared__ float red[256];
    const int o = blockIdx.x;
    const float* row = w + (size_t)o * I_DIM;
    float m = 0.0f;
    for (int i = threadIdx.x; i < I_DIM; i += 256)
        m = fmaxf(m, fabsf(row[i]));
    red[threadIdx.x] = m;
    __syncthreads();
    for (int s = 128; s > 0; s >>= 1) {
        if (threadIdx.x < s)
            red[threadIdx.x] = fmaxf(red[threadIdx.x], red[threadIdx.x + s]);
        __syncthreads();
    }
    if (threadIdx.x == 0)
        scale[o] = fmaxf(red[0], 1e-6f);
}

// ---------------------------------------------------------------------------
// Kernel 2: nearest-codeword quantization, dequantize directly to bf16.
// Codebook + ||c||^2 in LDS; all lanes scan the same codeword -> broadcast.
// First-min tie-break matches jnp.argmin.
// ---------------------------------------------------------------------------
__global__ __launch_bounds__(256)
void quantize_kernel(const float* __restrict__ w,
                     const float* __restrict__ cb,
                     const float* __restrict__ scale,
                     unsigned short* __restrict__ wq) {
    __shared__ float cbs[NCODE * PQ_D];   // 8 KB
    __shared__ float c2s[NCODE];          // 1 KB
    const int tid = threadIdx.x;

    float4 c0 = *(const float4*)(&cb[tid * PQ_D]);
    float4 c1 = *(const float4*)(&cb[tid * PQ_D + 4]);
    *(float4*)(&cbs[tid * PQ_D])     = c0;
    *(float4*)(&cbs[tid * PQ_D + 4]) = c1;
    c2s[tid] = c0.x*c0.x + c0.y*c0.y + c0.z*c0.z + c0.w*c0.w +
               c1.x*c1.x + c1.y*c1.y + c1.z*c1.z + c1.w*c1.w;
    __syncthreads();

    const size_t gid = (size_t)blockIdx.x * 256 + tid;       // group index
    const int o = (int)(gid >> 9);                            // 512 groups / row
    const float s = scale[o];

    float4 w0 = *(const float4*)(&w[gid * PQ_D]);
    float4 w1 = *(const float4*)(&w[gid * PQ_D + 4]);
    float g[PQ_D];
    g[0] = w0.x / s; g[1] = w0.y / s; g[2] = w0.z / s; g[3] = w0.w / s;
    g[4] = w1.x / s; g[5] = w1.y / s; g[6] = w1.z / s; g[7] = w1.w / s;

    int best = 0;
    float bd = 3.402823466e+38f;
    for (int k = 0; k < NCODE; ++k) {
        const float* c = &cbs[k * PQ_D];
        float dot = g[0] * c[0];
        #pragma unroll
        for (int j = 1; j < PQ_D; ++j) dot = fmaf(g[j], c[j], dot);
        float d = c2s[k] - 2.0f * dot;
        if (d < bd) { bd = d; best = k; }
    }

    const float* c = &cbs[best * PQ_D];
    ushort4 q0, q1;
    q0.x = f2bf(c[0] * s); q0.y = f2bf(c[1] * s);
    q0.z = f2bf(c[2] * s); q0.w = f2bf(c[3] * s);
    q1.x = f2bf(c[4] * s); q1.y = f2bf(c[5] * s);
    q1.z = f2bf(c[6] * s); q1.w = f2bf(c[7] * s);
    *(ushort4*)(&wq[gid * PQ_D])     = q0;
    *(ushort4*)(&wq[gid * PQ_D + 4]) = q1;
}

// ---------------------------------------------------------------------------
// Kernel 3: GEMM  out[M,N] = x[M,K] @ wq[N,K]^T + bias   via bf16 WMMA.
// 256 threads = 8 waves in 4(M) x 2(N); each wave owns 2x4 subtiles of 16x16,
// 8 x v_wmma_f32_16x16x32_bf16 per K-step. Double-buffered LDS: next B tile
// streams in via GLOBAL_LOAD_ASYNC_TO_LDS_B128 and next A tile is held in
// registers (f32->bf16 converted) while the current tile's WMMAs execute.
// ---------------------------------------------------------------------------
union FragAB {
    v16bf v;
    struct { uint4 lo; uint4 hi; } q;
};

__global__ __launch_bounds__(256)
void gemm_kernel(const float* __restrict__ x,
                 const unsigned short* __restrict__ wq,
                 const float* __restrict__ bias,
                 float* __restrict__ out) {
    __shared__ unsigned short As[2][BM * LDT];   // 2 x 10 KB (x tile, bf16)
    __shared__ unsigned short Bs[2][BN * LDT];   // 2 x 10 KB (wq tile, bf16)

    const int tid  = threadIdx.x;
    const int lane = tid & 31;
    const int wv   = tid >> 5;      // wave 0..7
    const int wm   = wv & 3;        // 4 waves along M
    const int wn   = wv >> 2;       // 2 waves along N

    const int block_m = blockIdx.y * BM;
    const int block_n = blockIdx.x * BN;

    const v8f vzero = {0.f, 0.f, 0.f, 0.f, 0.f, 0.f, 0.f, 0.f};
    v8f acc[2][4];
    #pragma unroll
    for (int i = 0; i < 2; ++i)
        #pragma unroll
        for (int j = 0; j < 4; ++j) acc[i][j] = vzero;

    // per-lane fragment addressing (ISA 7.12.2, 16-bit layouts)
    const int lrow = lane & 15;
    const int a_k0 = (lane < 16) ? 0 : 8;    // A: K halves split across lane halves
    const int b_k0 = (lane < 16) ? 0 : 8;    // B: K halves split across lane halves (same as A)

    // staging decomposition
    const int ar  = tid >> 3;          // 0..31  (x rows, 4 passes of 32)
    const int ac4 = (tid & 7) * 4;     // f32 quad within BK
    const int br  = tid >> 2;          // 0..63  (wq rows, 2 passes of 64)
    const int bc8 = (tid & 3) * 8;     // bf16 octet within BK

    const float*          xrow = x  + (size_t)(block_m + ar) * K_DIM + ac4;
    const unsigned short* wrow = wq + (size_t)(block_n + br) * K_DIM + bc8;

    float4 areg[4];

    // issue next tile: async B -> LDS[buf], A f32 -> registers
    auto issue_tile = [&](int buf, int k0) {
        #pragma unroll
        for (int p = 0; p < 2; ++p)
            stage_b16(&Bs[buf][(br + p * 64) * LDT + bc8],
                      wrow + (size_t)p * 64 * K_DIM + k0);
        #pragma unroll
        for (int p = 0; p < 4; ++p)
            areg[p] = *(const float4*)(xrow + (size_t)p * 32 * K_DIM + k0);
    };
    // convert held A registers and commit to LDS[buf]
    auto commit_a = [&](int buf) {
        #pragma unroll
        for (int p = 0; p < 4; ++p) {
            ushort4 h;
            h.x = f2bf(areg[p].x); h.y = f2bf(areg[p].y);
            h.z = f2bf(areg[p].z); h.w = f2bf(areg[p].w);
            *(ushort4*)(&As[buf][(ar + p * 32) * LDT + ac4]) = h;
        }
    };

    // prologue: fill buffer 0
    issue_tile(0, 0);
    commit_a(0);
    wait_async_lds();
    __syncthreads();

    for (int kt = 0; kt < NKIT; ++kt) {
        const int  cur  = kt & 1;
        const int  nxt  = cur ^ 1;
        const bool more = (kt + 1) < NKIT;

        if (more) issue_tile(nxt, (kt + 1) * BK);   // overlaps with WMMAs below

        FragAB a[2], b[4];
        #pragma unroll
        for (int i = 0; i < 2; ++i) {
            const int row = wm * 32 + i * 16 + lrow;
            a[i].q.lo = *(const uint4*)(&As[cur][row * LDT + a_k0]);
            a[i].q.hi = *(const uint4*)(&As[cur][row * LDT + 16 + a_k0]);
        }
        #pragma unroll
        for (int j = 0; j < 4; ++j) {
            const int row = wn * 64 + j * 16 + lrow;
            b[j].q.lo = *(const uint4*)(&Bs[cur][row * LDT + b_k0]);
            b[j].q.hi = *(const uint4*)(&Bs[cur][row * LDT + 16 + b_k0]);
        }

        #pragma unroll
        for (int i = 0; i < 2; ++i)
            #pragma unroll
            for (int j = 0; j < 4; ++j)
                acc[i][j] = __builtin_amdgcn_wmma_f32_16x16x32_bf16(
                    false, a[i].v, false, b[j].v, (short)0, acc[i][j], false, false);

        if (more) commit_a(nxt);
        wait_async_lds();          // this wave's async B portion landed
        __syncthreads();           // all waves' A commits + B asyncs visible
    }

    // epilogue: C/D layout (lanes 0-15: rows m0+r; 16-31: rows m0+8+r)
    const int rbase_off = (lane >> 4) * 8;
    #pragma unroll
    for (int i = 0; i < 2; ++i) {
        const int m0 = block_m + wm * 32 + i * 16 + rbase_off;
        #pragma unroll
        for (int j = 0; j < 4; ++j) {
            const int col = block_n + wn * 64 + j * 16 + (lane & 15);
            const float bv = bias[col];
            #pragma unroll
            for (int r = 0; r < 8; ++r)
                out[(size_t)(m0 + r) * N_DIM + col] = acc[i][j][r] + bv;
        }
    }
}

// ---------------------------------------------------------------------------
// Launch: row scales -> quantize (dequant to bf16 in ws) -> WMMA GEMM.
// Workspace: [0, 8KB) row_scale f32; [8KB, 8KB+16MB) wq bf16.
// ---------------------------------------------------------------------------
extern "C" void kernel_launch(void* const* d_in, const int* in_sizes, int n_in,
                              void* d_out, int out_size, void* d_ws, size_t ws_size,
                              hipStream_t stream) {
    const float* x        = (const float*)d_in[0];
    const float* weight   = (const float*)d_in[1];
    const float* codebook = (const float*)d_in[2];
    const float* bias     = (const float*)d_in[3];
    float*       out      = (float*)d_out;

    float*          row_scale = (float*)d_ws;
    unsigned short* wq        = (unsigned short*)((char*)d_ws + 8192);

    rowscale_kernel<<<O_DIM, 256, 0, stream>>>(weight, row_scale);

    const int ngroups = (O_DIM * I_DIM) / PQ_D;            // 1,048,576
    quantize_kernel<<<ngroups / 256, 256, 0, stream>>>(weight, codebook, row_scale, wq);

    dim3 grid(N_DIM / BN, M_DIM / BM);                     // (16, 64)
    gemm_kernel<<<grid, 256, 0, stream>>>(x, wq, bias, out);
}
